// GNNPolicy_19713899889087
// MI455X (gfx1250) — matrix operand order repaired and checked by reading, not compiled, not run on hardware
//
#include <hip/hip_runtime.h>

// ---------------------------------------------------------------------------
// CDNA5 (gfx1250) GNN forward. wave32. WMMA bf16 16x16x32, f32 accumulate.
// t_tgt/t_src stored bf16 (halves the dominant gather traffic); LN + segment
// sum in f32.
// ---------------------------------------------------------------------------

typedef __attribute__((ext_vector_type(16))) __bf16   v16bf;
typedef __attribute__((ext_vector_type(8)))  float    v8f;
typedef __attribute__((ext_vector_type(8)))  unsigned v8u;

// round-to-nearest-even f32 -> bf16 bits in [31:16]
__device__ __forceinline__ unsigned rne(float f) {
  unsigned u = __builtin_bit_cast(unsigned, f);
  return u + 0x7FFFu + ((u >> 16) & 1u);
}

// pack two f32 into one VGPR of two bf16 (low = a, high = b)
__device__ __forceinline__ unsigned pkbf(float a, float b) {
#if __has_builtin(__builtin_amdgcn_cvt_pk_bf16_f32)
  typedef __attribute__((ext_vector_type(2))) __bf16 v2bf;
  v2bf p = __builtin_amdgcn_cvt_pk_bf16_f32(a, b);
  return __builtin_bit_cast(unsigned, p);
#elif __has_builtin(__builtin_amdgcn_perm)
  // dst = [a.b2, a.b3, b.b2, b.b3] : one v_perm_b32 merge
  return __builtin_amdgcn_perm(rne(b), rne(a), 0x07060302u);
#else
  return (rne(a) >> 16) | (rne(b) & 0xffff0000u);
#endif
}

// unpack bf16 pair word -> f32
__device__ __forceinline__ float bflo(unsigned u) {
  return __builtin_bit_cast(float, u << 16);
}
__device__ __forceinline__ float bfhi(unsigned u) {
  return __builtin_bit_cast(float, u & 0xffff0000u);
}

__device__ __forceinline__ v8f wmma_bf16(v16bf a, v16bf b, v8f c) {
  // (neg_a, A, neg_b, B, c_mod, C, reuse_a, reuse_b)
  return __builtin_amdgcn_wmma_f32_16x16x32_bf16(false, a, false, b, (short)0, c,
                                                 false, false);
}

// CDNA5 16-bit A-matrix 16x32 layout, expressed in k-PAIRS (pairs contiguous):
// VGPR p holds pair (p<4 ? p : p+4), lanes 16-31 add +4 pairs. Row M = lane&15.
__device__ __forceinline__ int a_pairmap(int p, int lane) {
  return ((p < 4) ? p : (p + 4)) + ((lane & 16) ? 4 : 0);
}
// 16-bit B-matrix 32x16 layout in k-pairs: same as A, pair (p<4 ? p : p+4) (+4 for lanes 16-31).
__device__ __forceinline__ int b_pairmap(int p, int lane) {
  return ((p < 4) ? p : (p + 4)) + ((lane & 16) ? 4 : 0);
}

// ---------------------------------------------------------------------------
// Generic node linear: Y[N,64] = relu?( X @ W[K,64] + bias ) (+ addv)
// K = K32*32 (64 or 128); K=128 takes concat [X1 | X2]. OUT16=1 stores packed
// bf16 (b16 stores) for the edge-gather t-buffers.
// ---------------------------------------------------------------------------
template <int K32, int OUT16>
__launch_bounds__(128)
__global__ void lin_wmma_kernel(const float* __restrict__ X1,
                                const float* __restrict__ X2,
                                const float* __restrict__ W,
                                const float* __restrict__ bias, int hasBias,
                                const float* __restrict__ addv, int hasAdd,
                                int doRelu, void* __restrict__ Yout, int N) {
  __shared__ unsigned Wp[K32 * 16 * 64];  // packed bf16 pairs [k2][n]
  __shared__ float xs[4][16][K32 * 32 + 8];
  const int tid = threadIdx.x, lane = tid & 31, w = tid >> 5;
  const int n16 = lane & 15;

  for (int i = tid; i < K32 * 16 * 64; i += 128) {
    int k2 = i >> 6, n = i & 63;
    Wp[i] = pkbf(W[(2 * k2) * 64 + n], W[(2 * k2 + 1) * 64 + n]);
  }
  __syncthreads();

  // loop-invariant B fragments: [n-tile][k-chunk]
  v16bf bf[4][K32];
#pragma unroll
  for (int t = 0; t < 4; ++t) {
#pragma unroll
    for (int c = 0; c < K32; ++c) {
      v8u tmp;
#pragma unroll
      for (int p = 0; p < 8; ++p)
        tmp[p] = Wp[(c * 16 + b_pairmap(p, lane)) * 64 + t * 16 + n16];
      bf[t][c] = __builtin_bit_cast(v16bf, tmp);
    }
  }

  const int ntiles = (N + 15) >> 4;
  for (int tile = blockIdx.x * 4 + w; tile < ntiles; tile += gridDim.x * 4) {
    const int m0 = tile << 4;
    // stage the 16 input rows into LDS (coalesced b128 loads, 32-bit offsets)
    int lr = lane >> 1;
    int grow = m0 + lr;
    if (grow >= N) grow = N - 1;
    const int d0 = (lane & 1) * 32;
    {
      const float4* p1 = (const float4*)(X1 + (unsigned)(grow * 64 + d0));
#pragma unroll
      for (int q = 0; q < 8; ++q) {
        float4 v4 = p1[q];
        int dd = d0 + 4 * q;
        xs[w][lr][dd + 0] = v4.x; xs[w][lr][dd + 1] = v4.y;
        xs[w][lr][dd + 2] = v4.z; xs[w][lr][dd + 3] = v4.w;
      }
      if constexpr (K32 == 4) {
        const float4* p2 = (const float4*)(X2 + (unsigned)(grow * 64 + d0));
#pragma unroll
        for (int q = 0; q < 8; ++q) {
          float4 v4 = p2[q];
          int dd = 64 + d0 + 4 * q;
          xs[w][lr][dd + 0] = v4.x; xs[w][lr][dd + 1] = v4.y;
          xs[w][lr][dd + 2] = v4.z; xs[w][lr][dd + 3] = v4.w;
        }
      }
    }
    __builtin_amdgcn_wave_barrier();  // LDS in-order per wave; fence compiler

    // A fragments: contiguous float2 reads + packed conversion
    v16bf af[K32];
#pragma unroll
    for (int c = 0; c < K32; ++c) {
      v8u tmp;
#pragma unroll
      for (int p = 0; p < 8; ++p) {
        int k = (c * 16 + a_pairmap(p, lane)) * 2;
        float2 xv = *(const float2*)&xs[w][n16][k];
        tmp[p] = pkbf(xv.x, xv.y);
      }
      af[c] = __builtin_bit_cast(v16bf, tmp);
    }

#pragma unroll
    for (int t = 0; t < 4; ++t) {
      v8f acc = {};
#pragma unroll
      for (int c = 0; c < K32; ++c) acc = wmma_bf16(af[c], bf[t][c], acc);
      const int n = t * 16 + n16;
      const float bv = hasBias ? bias[n] : 0.0f;
#pragma unroll
      for (int r = 0; r < 8; ++r) {
        int mrow = m0 + r + ((lane & 16) ? 8 : 0);
        if (mrow < N) {
          unsigned off = (unsigned)(mrow * 64 + n);
          float val = acc[r] + bv;
          if (doRelu) val = fmaxf(val, 0.0f);
          if (hasAdd) val += addv[off];
          if constexpr (OUT16) {
            ((unsigned short*)Yout)[off] = (unsigned short)(rne(val) >> 16);
          } else {
            ((float*)Yout)[off] = val;
          }
        }
      }
    }
    __builtin_amdgcn_wave_barrier();
  }
}

// ---------------------------------------------------------------------------
// Edge conv: per wave, 16 edges. tt/ts are bf16 rows (packed pairs, 32 uints
// per row). z = tt[dst] + ts[src] + te unpacked to f32 in LDS; per-edge LN +
// relu; bf16 WMMA against wf (B hoisted, pre-packed); f32 atomic scatter-add
// of (acc + bf) into agg[dst]. te[d] = edge_ln_b0 * cv_we[i][d].
// ---------------------------------------------------------------------------
__launch_bounds__(256)
__global__ void edge_conv_kernel(const unsigned* __restrict__ tt,
                                 const unsigned* __restrict__ ts,
                                 const int* __restrict__ dstIdx,
                                 const int* __restrict__ srcIdx, int E,
                                 const float* __restrict__ wf,
                                 const float* __restrict__ bfp,
                                 const float* __restrict__ lng,
                                 const float* __restrict__ lnb,
                                 const float* __restrict__ we,
                                 const float* __restrict__ elnb,
                                 float* __restrict__ agg) {
  __shared__ unsigned Wp[32 * 64];  // wf packed bf16 pairs [k2][n]
  __shared__ float zs[8][16][72];
  __shared__ int dstl[8][16];
  __shared__ float telds[64], lngl[64], lnbl[64], bfl[64];
  const int tid = threadIdx.x, lane = tid & 31, w = tid >> 5;
  const int n16 = lane & 15;

  for (int i = tid; i < 32 * 64; i += 256) {
    int k2 = i >> 6, n = i & 63;
    Wp[i] = pkbf(wf[(2 * k2) * 64 + n], wf[(2 * k2 + 1) * 64 + n]);
  }
  if (tid < 64) {
    telds[tid] = elnb[0] * we[tid];  // LN over size-1 axis collapses to beta
    lngl[tid] = lng[tid];
    lnbl[tid] = lnb[tid];
    bfl[tid] = bfp[tid];
  }
  __syncthreads();

  // hoisted loop-invariant B fragments of wf
  v16bf bfr[4][2];
#pragma unroll
  for (int t = 0; t < 4; ++t) {
#pragma unroll
    for (int c = 0; c < 2; ++c) {
      v8u tmp;
#pragma unroll
      for (int p = 0; p < 8; ++p)
        tmp[p] = Wp[(c * 16 + b_pairmap(p, lane)) * 64 + t * 16 + n16];
      bfr[t][c] = __builtin_bit_cast(v16bf, tmp);
    }
  }

  const int ntiles = E >> 4;  // E is a multiple of 16
  for (int tile = blockIdx.x * 8 + w; tile < ntiles; tile += gridDim.x * 8) {
    const int e0 = tile << 4;
    int nxt = tile + (int)gridDim.x * 8;
    if (nxt < ntiles) {  // speculative prefetch of next index block
      __builtin_prefetch(dstIdx + (nxt << 4), 0, 1);
      __builtin_prefetch(srcIdx + (nxt << 4), 0, 1);
    }

    const int erow = lane >> 1;
    const int d0 = (lane & 1) * 32;          // 32 dims handled by this lane
    const int el = e0 + erow;
    const int di = dstIdx[el];
    const int si = srcIdx[el];
    if ((lane & 1) == 0) dstl[w][erow] = di;

    // gather bf16 rows: 4 x b128 per table per lane (vs 8 for f32)
    const uint4* pt = (const uint4*)(tt + (unsigned)(di * 32 + (lane & 1) * 16));
    const uint4* ps = (const uint4*)(ts + (unsigned)(si * 32 + (lane & 1) * 16));
#pragma unroll
    for (int q = 0; q < 4; ++q) {
      uint4 ta = pt[q], sa = ps[q];
      int dd = d0 + q * 8;
      zs[w][erow][dd + 0] = bflo(ta.x) + bflo(sa.x) + telds[dd + 0];
      zs[w][erow][dd + 1] = bfhi(ta.x) + bfhi(sa.x) + telds[dd + 1];
      zs[w][erow][dd + 2] = bflo(ta.y) + bflo(sa.y) + telds[dd + 2];
      zs[w][erow][dd + 3] = bfhi(ta.y) + bfhi(sa.y) + telds[dd + 3];
      zs[w][erow][dd + 4] = bflo(ta.z) + bflo(sa.z) + telds[dd + 4];
      zs[w][erow][dd + 5] = bfhi(ta.z) + bfhi(sa.z) + telds[dd + 5];
      zs[w][erow][dd + 6] = bflo(ta.w) + bflo(sa.w) + telds[dd + 6];
      zs[w][erow][dd + 7] = bfhi(ta.w) + bfhi(sa.w) + telds[dd + 7];
    }
    __builtin_amdgcn_wave_barrier();

    // per-edge LN stats: lane pair (l, l+16) each sums 32 values of row l&15
    float s = 0.f, s2 = 0.f;
    {
      const float4* zr = (const float4*)&zs[w][n16][(lane & 16) ? 32 : 0];
#pragma unroll
      for (int q = 0; q < 8; ++q) {
        float4 xv = zr[q];
        s += xv.x + xv.y + xv.z + xv.w;
        s2 += xv.x * xv.x + xv.y * xv.y + xv.z * xv.z + xv.w * xv.w;
      }
    }
    s += __shfl_xor(s, 16, 32);
    s2 += __shfl_xor(s2, 16, 32);
    const float mu = s * (1.0f / 64.0f);
    const float var = s2 * (1.0f / 64.0f) - mu * mu;
    const float rs = rsqrtf(var + 1e-5f);

    // A fragments: LN + relu fused into packed conversion
    v16bf af[2];
#pragma unroll
    for (int c = 0; c < 2; ++c) {
      v8u tmp;
#pragma unroll
      for (int p = 0; p < 8; ++p) {
        int k = (c * 16 + a_pairmap(p, lane)) * 2;
        float2 xv = *(const float2*)&zs[w][n16][k];
        float x0 = fmaxf((xv.x - mu) * rs * lngl[k] + lnbl[k], 0.0f);
        float x1 = fmaxf((xv.y - mu) * rs * lngl[k + 1] + lnbl[k + 1], 0.0f);
        tmp[p] = pkbf(x0, x1);
      }
      af[c] = __builtin_bit_cast(v16bf, tmp);
    }

#pragma unroll
    for (int t = 0; t < 4; ++t) {
      v8f acc = {};
      acc = wmma_bf16(af[0], bfr[t][0], acc);
      acc = wmma_bf16(af[1], bfr[t][1], acc);
      const int n = t * 16 + n16;
      const float bias = bfl[n];
#pragma unroll
      for (int r = 0; r < 8; ++r) {
        int mloc = r + ((lane & 16) ? 8 : 0);
        unsigned off = (unsigned)(dstl[w][mloc] * 64 + n);
        unsafeAtomicAdd(agg + off, acc[r] + bias);
      }
    }
    __builtin_amdgcn_wave_barrier();
  }
}

// ---------------------------------------------------------------------------
// LayerNorm over last dim (64): wave per row, shfl reductions.
// ---------------------------------------------------------------------------
__launch_bounds__(256)
__global__ void ln64_kernel(const float* __restrict__ X,
                            const float* __restrict__ g,
                            const float* __restrict__ b,
                            float* __restrict__ Y, int N) {
  const int lane = threadIdx.x & 31, w = threadIdx.x >> 5;
  for (int r = blockIdx.x * 8 + w; r < N; r += gridDim.x * 8) {
    const float* xr = X + (unsigned)(r * 64);
    float x0 = xr[lane], x1 = xr[lane + 32];
    float s = x0 + x1;
#pragma unroll
    for (int off = 16; off > 0; off >>= 1) s += __shfl_xor(s, off, 32);
    float m = s * (1.0f / 64.0f);
    float d0 = x0 - m, d1 = x1 - m;
    float v = d0 * d0 + d1 * d1;
#pragma unroll
    for (int off = 16; off > 0; off >>= 1) v += __shfl_xor(v, off, 32);
    float rs = rsqrtf(v * (1.0f / 64.0f) + 1e-5f);
    float* yr = Y + (unsigned)(r * 64);
    yr[lane] = d0 * rs * g[lane] + b[lane];
    yr[lane + 32] = d1 * rs * g[lane + 32] + b[lane + 32];
  }
}

// ---------------------------------------------------------------------------
// Front embeddings: Y[N,64] = relu( LN_K(x) @ W1[K,64] + b1 ), K in {2,7,16}.
// ---------------------------------------------------------------------------
__global__ void front_ln_lin_kernel(const float* __restrict__ x, int K,
                                    const float* __restrict__ lng,
                                    const float* __restrict__ lnb,
                                    const float* __restrict__ w1,
                                    const float* __restrict__ b1,
                                    float* __restrict__ Y, int N) {
  const long long total = (long long)N * 64;
  for (long long i = (long long)blockIdx.x * blockDim.x + threadIdx.x; i < total;
       i += (long long)gridDim.x * blockDim.x) {
    int node = (int)(i >> 6), d = (int)(i & 63);
    const float* xr = x + (long long)node * K;
    float s = 0.f, s2 = 0.f;
    for (int k = 0; k < K; ++k) { float xx = xr[k]; s += xx; s2 += xx * xx; }
    float inv = 1.0f / (float)K;
    float m = s * inv;
    float var = s2 * inv - m * m;
    float rs = rsqrtf(var + 1e-5f);
    float acc = b1[d];
    for (int k = 0; k < K; ++k) {
      float xn = (xr[k] - m) * rs * lng[k] + lnb[k];
      acc += xn * w1[k * 64 + d];
    }
    Y[i] = fmaxf(acc, 0.0f);
  }
}

__global__ void head_dot_kernel(const float* __restrict__ H,
                                const float* __restrict__ w2,
                                float* __restrict__ out, int N) {
  for (int i = blockIdx.x * blockDim.x + threadIdx.x; i < N;
       i += gridDim.x * blockDim.x) {
    const float4* p = (const float4*)(H + (unsigned)(i * 64));
    float s = 0.f;
#pragma unroll
    for (int q = 0; q < 16; ++q) {
      float4 v = p[q];
      s += v.x * w2[4 * q] + v.y * w2[4 * q + 1] + v.z * w2[4 * q + 2] +
           v.w * w2[4 * q + 3];
    }
    out[i] = s;
  }
}

__global__ void zero_kernel(float4* __restrict__ p, long long n4) {
  float4 z = {0.f, 0.f, 0.f, 0.f};
  for (long long i = (long long)blockIdx.x * blockDim.x + threadIdx.x; i < n4;
       i += (long long)gridDim.x * blockDim.x)
    p[i] = z;
}

// ---------------------------------------------------------------------------
// Orchestration
// ---------------------------------------------------------------------------
extern "C" void kernel_launch(void* const* d_in, const int* in_sizes, int n_in,
                              void* d_out, int out_size, void* d_ws,
                              size_t ws_size, hipStream_t stream) {
  const float* cons_x    = (const float*)d_in[0];
  /* d_in[1] edge_x: dead (LN over size-1 axis == beta) */
  const float* var_x     = (const float*)d_in[2];
  const float* grp_x     = (const float*)d_in[3];
  const int*   edge_idx  = (const int*)d_in[4];
  const float* grp_ln_g  = (const float*)d_in[5];
  const float* grp_ln_b  = (const float*)d_in[6];
  const float* grp_w     = (const float*)d_in[7];
  const float* grp_b     = (const float*)d_in[8];
  const float* cons_ln_g = (const float*)d_in[9];
  const float* cons_ln_b = (const float*)d_in[10];
  const float* cons_w1   = (const float*)d_in[11];
  const float* cons_b1   = (const float*)d_in[12];
  const float* cons_w2   = (const float*)d_in[13];
  const float* cons_b2   = (const float*)d_in[14];
  /* d_in[15] edge_ln_g: dead */
  const float* edge_ln_b = (const float*)d_in[16];
  const float* var_ln_g  = (const float*)d_in[17];
  const float* var_ln_b  = (const float*)d_in[18];
  const float* var_w1    = (const float*)d_in[19];
  const float* var_b1    = (const float*)d_in[20];
  const float* var_w2    = (const float*)d_in[21];
  const float* var_b2    = (const float*)d_in[22];
  const float* cv_wl     = (const float*)d_in[23];
  const float* cv_bl     = (const float*)d_in[24];
  const float* cv_we     = (const float*)d_in[25];
  const float* cv_wr     = (const float*)d_in[26];
  const float* cv_lnf_g  = (const float*)d_in[27];
  const float* cv_lnf_b  = (const float*)d_in[28];
  const float* cv_wf     = (const float*)d_in[29];
  const float* cv_bf     = (const float*)d_in[30];
  const float* cv_post_g = (const float*)d_in[31];
  const float* cv_post_b = (const float*)d_in[32];
  const float* cv_wo1    = (const float*)d_in[33];
  const float* cv_bo1    = (const float*)d_in[34];
  const float* cv_wo2    = (const float*)d_in[35];
  const float* cv_bo2    = (const float*)d_in[36];
  const float* out_w1    = (const float*)d_in[37];
  const float* out_b1    = (const float*)d_in[38];
  const float* out_w2    = (const float*)d_in[39];

  const int C = in_sizes[0] / 2;
  const int V = in_sizes[2] / 7;
  const int E = in_sizes[4] / 2;
  const int NM = (C > V) ? C : V;
  const size_t NB = (size_t)NM * 64;

  float* ws = (float*)d_ws;
  float* b_c   = ws + 0 * NB;
  float* b_v   = ws + 1 * NB;
  float* b_tt  = ws + 2 * NB;  // t_tgt: bf16 (uses half the slab); also f32 tmp
  float* b_ts  = ws + 3 * NB;  // t_src: bf16 / f32 tmp
  float* b_agg = ws + 4 * NB;
  float* b_ln  = ws + 5 * NB;
  float* b_y1  = ws + 6 * NB;

  const int LIN_BLOCKS = 768;    // 4 row-tiles per block per pass
  const int EDGE_BLOCKS = 2048;  // 8 edge-tiles per block per pass

  auto lin64 = [&](const float* X, const float* W, const float* bias,
                   const float* addv, int relu, float* Y, int N) {
    lin_wmma_kernel<2, 0><<<LIN_BLOCKS, 128, 0, stream>>>(
        X, nullptr, W, bias, bias ? 1 : 0, addv, addv ? 1 : 0, relu, Y, N);
  };
  auto lin64_bf = [&](const float* X, const float* W, const float* bias,
                      float* Y, int N) {  // bf16 output for t-buffers
    lin_wmma_kernel<2, 1><<<LIN_BLOCKS, 128, 0, stream>>>(
        X, nullptr, W, bias, bias ? 1 : 0, nullptr, 0, 0, Y, N);
  };
  auto lin128 = [&](const float* X1, const float* X2, const float* W,
                    const float* bias, int relu, float* Y, int N) {
    lin_wmma_kernel<4, 0><<<LIN_BLOCKS, 128, 0, stream>>>(X1, X2, W, bias, 1,
                                                          nullptr, 0, relu, Y, N);
  };

  // ---- embeddings ----
  {
    long long tc = (long long)C * 64, tv = (long long)V * 64;
    front_ln_lin_kernel<<<(unsigned)((tc + 255) / 256), 256, 0, stream>>>(
        cons_x, 2, cons_ln_g, cons_ln_b, cons_w1, cons_b1, b_tt, C);
    front_ln_lin_kernel<<<(unsigned)((tv + 255) / 256), 256, 0, stream>>>(
        var_x, 7, var_ln_g, var_ln_b, var_w1, var_b1, b_ts, V);
    front_ln_lin_kernel<<<(unsigned)((tv + 255) / 256), 256, 0, stream>>>(
        grp_x, 16, grp_ln_g, grp_ln_b, grp_w, grp_b, b_agg, V);
  }
  lin64(b_tt, cons_w2, cons_b2, nullptr, 1, b_c, C);  // c
  lin64(b_ts, var_w2, var_b2, b_agg, 1, b_v, V);      // v = relu(..) + g

  // ---- 4 bipartite convs ----
  const int* ci = edge_idx;
  const int* vi = edge_idx + E;
  for (int i = 0; i < 4; ++i) {
    const bool isC = (i % 2 == 0);  // convs 0,2: right=c ; convs 1,3: right=v
    float* right = isC ? b_c : b_v;
    float* left  = isC ? b_v : b_c;
    const int* dsts = isC ? ci : vi;
    const int* srcs = isC ? vi : ci;
    const int NR = isC ? C : V;
    const int NL = isC ? V : C;

    lin64_bf(right, cv_wl + i * 4096, cv_bl + i * 64, b_tt, NR);  // t_tgt bf16
    lin64_bf(left,  cv_wr + i * 4096, nullptr,        b_ts, NL);  // t_src bf16
    zero_kernel<<<1024, 256, 0, stream>>>((float4*)b_agg, (long long)NR * 16);
    edge_conv_kernel<<<EDGE_BLOCKS, 256, 0, stream>>>(
        (const unsigned*)b_tt, (const unsigned*)b_ts, dsts, srcs, E,
        cv_wf + i * 4096, cv_bf + i * 64, cv_lnf_g + i * 64, cv_lnf_b + i * 64,
        cv_we + i * 64, edge_ln_b, b_agg);
    ln64_kernel<<<2048, 256, 0, stream>>>(b_agg, cv_post_g + i * 64,
                                          cv_post_b + i * 64, b_ln, NR);
    lin128(b_ln, right, cv_wo1 + i * 8192, cv_bo1 + i * 64, 1, b_y1, NR);
    lin64(b_y1, cv_wo2 + i * 4096, cv_bo2 + i * 64, nullptr, 0, right, NR);
  }

  // ---- head ----
  lin64(b_v, out_w1, out_b1, nullptr, 1, b_tt, V);
  head_dot_kernel<<<(V + 255) / 256, 256, 0, stream>>>(b_tt, out_w2,
                                                       (float*)d_out, V);
}
